// RASNet_Block_28965259444684
// MI455X (gfx1250) — matrix operand order repaired and checked by reading, not compiled, not run on hardware
//
#include <hip/hip_runtime.h>
#include <math.h>

typedef _Float16 f16;
typedef _Float16 v16h __attribute__((ext_vector_type(16)));
typedef float    v8f  __attribute__((ext_vector_type(8)));

#define BB 4
#define CC 128
#define HH 48
#define WW 48
#define NN 2304
#define DC 256

enum { EPI_NONE = 0, EPI_SIGMOID = 1, EPI_RELU = 2, EPI_RELUSQ = 3, EPI_ADD = 4 };

// ---------------------------------------------------------------------------
// WMMA GEMM: C[M,N] = epilogue(scale * A·B + bias[, + aux])
// A: M×K (TA=0) or K×M storage (TA=1).  B: K×N (TB=0) or N×K storage (TB=1).
// Block = 256 threads = 8 wave32; tile 128(M) × 64(N) × 32(K), double-buffered
// LDS. Each wave computes a 32×32 subtile = 2×2 v_wmma_f32_16x16x32_f16.
// All K are multiples of 32 and all N multiples of 64 in this net; M may be
// partial (handled branchlessly with clamped addresses + zero mask).
// TA=1 is only used with M=2304 (always full tiles).
// ---------------------------------------------------------------------------
template<int TA, int TB, int EPI>
__global__ void __launch_bounds__(256)
gemm_wmma(const float* __restrict__ A, long long sAb,
          const float* __restrict__ B, long long sBb,
          float* __restrict__ C, long long sCb,
          const float* __restrict__ bias,
          const float* __restrict__ aux, long long sXb,
          int M, int N, int K, int lda, int ldb, int ldc, float scale)
{
  __shared__ __align__(16) f16 sA[2][128 * 32];
  __shared__ __align__(16) f16 sB[2][64 * 32];

  const int tid = threadIdx.x;
  const int m0 = blockIdx.y * 128;
  const int n0 = blockIdx.x * 64;
  A += (long long)blockIdx.z * sAb;
  B += (long long)blockIdx.z * sBb;
  C += (long long)blockIdx.z * sCb;
  if (EPI == EPI_ADD) aux += (long long)blockIdx.z * sXb;

  const int lane  = tid & 31;
  const int wv    = tid >> 5;
  const int wm    = (wv & 3) << 5;
  const int wn    = (wv >> 2) << 5;
  const int lhalf = lane >> 4;
  const int lrow  = lane & 15;
  const bool fullA = (m0 + 128 <= M);

  // Branchless vectorized tile staging (global float4 -> f16 LDS tiles).
  auto stage = [&](int buf, int kk) {
    f16* __restrict__ dA = sA[buf];
    f16* __restrict__ dB = sB[buf];
    if (TA == 0) {
      // 8 contiguous k per thread; 128x32 = 512 chunks -> 2 iters.
#pragma unroll
      for (int it = 0; it < 2; ++it) {
        int q = tid + (it << 8);
        int r = q >> 2;
        int c = (q & 3) << 3;
        int gm = m0 + r;
        int gmc = fullA ? gm : (gm < M ? gm : M - 1);
        float z  = fullA ? 1.f : (gm < M ? 1.f : 0.f);
        const float* Ap = A + (long long)gmc * lda + kk + c;
        float4 va = *(const float4*)Ap;
        float4 vb = *(const float4*)(Ap + 4);
        union { f16 h[8]; uint4 u; } p;
        p.h[0] = (f16)(va.x * z); p.h[1] = (f16)(va.y * z);
        p.h[2] = (f16)(va.z * z); p.h[3] = (f16)(va.w * z);
        p.h[4] = (f16)(vb.x * z); p.h[5] = (f16)(vb.y * z);
        p.h[6] = (f16)(vb.z * z); p.h[7] = (f16)(vb.w * z);
        *(uint4*)&dA[r * 32 + c] = p.u;
      }
    } else {
      // Transposed A (storage K×M): 4 contiguous m per thread, coalesced.
#pragma unroll
      for (int it = 0; it < 4; ++it) {
        int q = tid + (it << 8);
        int m = (q & 31) << 2;
        int k = q >> 5;
        float4 v4 = *(const float4*)(A + (long long)(kk + k) * lda + m0 + m);
        dA[(m + 0) * 32 + k] = (f16)v4.x;
        dA[(m + 1) * 32 + k] = (f16)v4.y;
        dA[(m + 2) * 32 + k] = (f16)v4.z;
        dA[(m + 3) * 32 + k] = (f16)v4.w;
      }
    }
    if (TB == 0) {
      // 4 contiguous n per thread; 32x64 = 512 chunks -> 2 iters.
#pragma unroll
      for (int it = 0; it < 2; ++it) {
        int q = tid + (it << 8);
        int n = (q & 15) << 2;
        int k = q >> 4;
        float4 v4 = *(const float4*)(B + (long long)(kk + k) * ldb + n0 + n);
        dB[(n + 0) * 32 + k] = (f16)v4.x;
        dB[(n + 1) * 32 + k] = (f16)v4.y;
        dB[(n + 2) * 32 + k] = (f16)v4.z;
        dB[(n + 3) * 32 + k] = (f16)v4.w;
      }
    } else {
      // Transposed B (storage N×K): 8 contiguous k per thread, one b128 store.
      int n = tid >> 2;
      int c = (tid & 3) << 3;
      const float* Bp = B + (long long)(n0 + n) * ldb + kk + c;
      float4 va = *(const float4*)Bp;
      float4 vb = *(const float4*)(Bp + 4);
      union { f16 h[8]; uint4 u; } p;
      p.h[0] = (f16)va.x; p.h[1] = (f16)va.y; p.h[2] = (f16)va.z; p.h[3] = (f16)va.w;
      p.h[4] = (f16)vb.x; p.h[5] = (f16)vb.y; p.h[6] = (f16)vb.z; p.h[7] = (f16)vb.w;
      *(uint4*)&dB[n * 32 + c] = p.u;
    }
  };

  const v8f vzero = {0.f, 0.f, 0.f, 0.f, 0.f, 0.f, 0.f, 0.f};
  v8f acc[2][2];
  acc[0][0] = vzero; acc[0][1] = vzero; acc[1][0] = vzero; acc[1][1] = vzero;

  const int nk = K >> 5;
  stage(0, 0);
  for (int kb = 0; kb < nk; ++kb) {
    __syncthreads();
    const int cur = kb & 1;
    // Fragment loads (CDNA5 16-bit layouts), issued before next-tile staging.
    v16h af[2], bf[2];
#pragma unroll
    for (int t = 0; t < 2; ++t) {
      const f16* pa = &sA[cur][(wm + t * 16 + lrow) * 32 + (lhalf << 3)];
      ((uint4*)&af[t])[0] = *(const uint4*)pa;
      ((uint4*)&af[t])[1] = *(const uint4*)(pa + 16);
      const f16* pb = &sB[cur][(wn + t * 16 + lrow) * 32 + (lhalf << 3)];
      ((uint4*)&bf[t])[0] = *(const uint4*)pb;
      ((uint4*)&bf[t])[1] = *(const uint4*)(pb + 16);
    }
    if (kb + 1 < nk) stage(cur ^ 1, (kb + 1) << 5);
#pragma unroll
    for (int tm = 0; tm < 2; ++tm)
#pragma unroll
      for (int tn = 0; tn < 2; ++tn)
        acc[tm][tn] = __builtin_amdgcn_wmma_f32_16x16x32_f16(
            false, af[tm], false, bf[tn], (short)0, acc[tm][tn], false, false);
  }

  // Epilogue + store (D layout: VGPR r -> M = r + lhalf*8, N = lane&15).
#pragma unroll
  for (int tm = 0; tm < 2; ++tm) {
#pragma unroll
    for (int tn = 0; tn < 2; ++tn) {
#pragma unroll
      for (int r = 0; r < 8; ++r) {
        int m = m0 + wm + tm * 16 + (lhalf << 3) + r;
        int n = n0 + wn + tn * 16 + lrow;
        if (m < M) {
          float y = acc[tm][tn][r] * scale;
          if (bias) y += bias[m];
          if (EPI == EPI_SIGMOID)     y = 1.f / (1.f + expf(-y));
          else if (EPI == EPI_RELU)   y = fmaxf(y, 0.f);
          else if (EPI == EPI_RELUSQ) { float t = fmaxf(y, 0.f); y = t * t; }
          else if (EPI == EPI_ADD)    y += aux[(long long)m * ldc + n];
          C[(long long)m * ldc + n] = y;
        }
      }
    }
  }
}

// ---------------------------------------------------------------------------
// Elementwise / reduction kernels
// ---------------------------------------------------------------------------
__global__ void rope_kernel(const float* __restrict__ x, float* __restrict__ y) {
  int idx = blockIdx.x * 256 + threadIdx.x;           // over B*64*N
  const int total = BB * 64 * NN;
  if (idx >= total) return;
  int n = idx % NN; int t = idx / NN; int p = t % 64; int b = t / 64;
  int h = n / WW, w = n % WW;
  float theta = powf(10000.f, -(float)p / 64.f);
  float pos = (float)(h + w) * theta;
  float cs = cosf(pos), sn = sinf(pos);
  const float* xb = x + ((size_t)b * CC + 2 * p) * NN + n;
  float re = xb[0], im = xb[NN];
  float* yb = y + ((size_t)b * CC + 2 * p) * NN + n;
  yb[0]  = re * cs - im * sn;
  yb[NN] = re * sn + im * cs;
}

__global__ void wkv_kernel(const float* __restrict__ k, const float* __restrict__ v,
                           const float* __restrict__ r, const float* __restrict__ u,
                           float* __restrict__ num, float* __restrict__ den,
                           float* __restrict__ rec) {
  int idx = blockIdx.x * 256 + threadIdx.x;
  const int total = BB * CC * NN;
  if (idx >= total) return;
  int c = (idx / NN) % CC;
  float kk = k[idx], vv = v[idx];
  float kv = kk * vv;
  num[idx] = kv;                     // new_num = 0*exp(-w_gate) + k*v
  den[idx] = kk;                     // new_den = 0*exp(-w_gate) + k
  float eu = expf(u[c]);
  float wkv = (kv + eu * kv) / (kk + eu * kk);
  rec[idx] = r[idx] * wkv;
}

__global__ void copy_x_to_feats(const float* __restrict__ x, float* __restrict__ feats) {
  int idx = blockIdx.x * 256 + threadIdx.x;
  const int total = BB * CC * NN;
  if (idx >= total) return;
  int b = idx / (CC * NN); int rest = idx % (CC * NN);
  feats[(size_t)b * DC * NN + rest] = x[idx];
}

__global__ void bn_stats(const float* __restrict__ feats, const float* __restrict__ g,
                         const float* __restrict__ bta, float* __restrict__ scale,
                         float* __restrict__ shift) {
  __shared__ float rs[256], rq[256];
  int c = blockIdx.x;
  float s = 0.f, q = 0.f;
  for (int i = threadIdx.x; i < BB * NN; i += 256) {
    int b = i / NN, n = i % NN;
    float v = feats[((size_t)b * DC + c) * NN + n];
    s += v; q += v * v;
  }
  rs[threadIdx.x] = s; rq[threadIdx.x] = q; __syncthreads();
  for (int o = 128; o > 0; o >>= 1) {
    if (threadIdx.x < o) { rs[threadIdx.x] += rs[threadIdx.x + o]; rq[threadIdx.x] += rq[threadIdx.x + o]; }
    __syncthreads();
  }
  if (threadIdx.x == 0) {
    const float inv = 1.f / (float)(BB * NN);
    float mean = rs[0] * inv;
    float var  = rq[0] * inv - mean * mean;
    float rstd = rsqrtf(var + 1e-5f);
    scale[c] = g[c] * rstd;
    shift[c] = bta[c] - mean * g[c] * rstd;
  }
}

__global__ void bn_apply_relu(const float* __restrict__ feats, const float* __restrict__ scale,
                              const float* __restrict__ shift, float* __restrict__ hbuf, int cin) {
  int idx = blockIdx.x * 256 + threadIdx.x;
  int total = BB * cin * NN;
  if (idx >= total) return;
  int n = idx % NN; int c = (idx / NN) % cin; int b = idx / (cin * NN);
  float v = feats[((size_t)b * DC + c) * NN + n];
  hbuf[idx] = fmaxf(v * scale[c] + shift[c], 0.f);
}

__global__ void conv3x3_dense(const float* __restrict__ hbuf, const float* __restrict__ wt,
                              const float* __restrict__ bias, float* __restrict__ feats,
                              int cin, int co0) {
  int idx = blockIdx.x * 256 + threadIdx.x;
  const int total = BB * 32 * NN;
  if (idx >= total) return;
  int n = idx % NN; int o = (idx / NN) % 32; int b = idx / (32 * NN);
  int hy = n / WW, wx = n % WW;
  float acc = bias[o];
  const float* ib = hbuf + (size_t)b * cin * NN;
  const float* wb = wt + (size_t)o * cin * 9;
  for (int c = 0; c < cin; ++c) {
    const float* ip = ib + (size_t)c * NN;
    const float* wp = wb + c * 9;
#pragma unroll
    for (int dy = -1; dy <= 1; ++dy) {
      int yy = hy + dy;
      if ((unsigned)yy >= HH) continue;
#pragma unroll
      for (int dx = -1; dx <= 1; ++dx) {
        int xx = wx + dx;
        if ((unsigned)xx >= WW) continue;
        acc += ip[yy * WW + xx] * wp[(dy + 1) * 3 + (dx + 1)];
      }
    }
  }
  feats[((size_t)b * DC + co0 + o) * NN + n] = acc;
}

__global__ void tau_decay(const float* __restrict__ s2, float* __restrict__ decay) {
  __shared__ float red[256];
  int bc = blockIdx.x;  // B*DC
  float s = 0.f;
  for (int i = threadIdx.x; i < NN; i += 256) s += s2[(size_t)bc * NN + i];
  red[threadIdx.x] = s; __syncthreads();
  for (int o = 128; o > 0; o >>= 1) {
    if (threadIdx.x < o) red[threadIdx.x] += red[threadIdx.x + o];
    __syncthreads();
  }
  if (threadIdx.x == 0) {
    float tau = red[0] / (float)NN;
    decay[bc] = expf(-1.f / (tau + 1e-6f));
  }
}

__global__ void spike_fuse(const float* __restrict__ df, const float* __restrict__ decay,
                           float* __restrict__ sf) {
  int idx = blockIdx.x * 256 + threadIdx.x;
  const int total = BB * DC * NN;
  if (idx >= total) return;
  int bc = idx / NN;
  float v = df[idx];
  float sp = 1.f / (1.f + expf(-(v - 1.f) * 5.f));
  sf[idx] = sp * v * decay[bc] + (1.f - sp) * v;
}

__global__ void softmax_rows(float* __restrict__ s) {
  __shared__ float red[256];
  float* row = s + (size_t)blockIdx.x * NN;
  int tid = threadIdx.x;
  float mx = -INFINITY;
  for (int i = tid; i < NN; i += 256) mx = fmaxf(mx, row[i]);
  red[tid] = mx; __syncthreads();
  for (int o = 128; o > 0; o >>= 1) { if (tid < o) red[tid] = fmaxf(red[tid], red[tid + o]); __syncthreads(); }
  mx = red[0]; __syncthreads();
  float sum = 0.f;
  for (int i = tid; i < NN; i += 256) { float e = expf(row[i] - mx); row[i] = e; sum += e; }
  red[tid] = sum; __syncthreads();
  for (int o = 128; o > 0; o >>= 1) { if (tid < o) red[tid] += red[tid + o]; __syncthreads(); }
  float inv = 1.f / red[0];
  for (int i = tid; i < NN; i += 256) row[i] *= inv;
}

// ---------------------------------------------------------------------------
// Orchestration
// ---------------------------------------------------------------------------
#define EWGRID(n) dim3(((n) + 255) / 256), dim3(256)

extern "C" void kernel_launch(void* const* d_in, const int* in_sizes, int n_in,
                              void* d_out, int out_size, void* d_ws, size_t ws_size,
                              hipStream_t stream) {
  (void)in_sizes; (void)n_in; (void)out_size; (void)ws_size;

  const float* x     = (const float*)d_in[0];
  const float* r_w   = (const float*)d_in[1];
  const float* r_b   = (const float*)d_in[2];
  const float* k_w   = (const float*)d_in[3];
  const float* k_b   = (const float*)d_in[4];
  const float* v_w   = (const float*)d_in[5];
  const float* v_b   = (const float*)d_in[6];
  // d_in[7..10] = wc1_w/wc1_b/wc2_w/wc2_b: gate multiplies zeros -> dead code.
  const float* u     = (const float*)d_in[11];
  const float* db_g[4]  = {(const float*)d_in[12], (const float*)d_in[16], (const float*)d_in[20], (const float*)d_in[24]};
  const float* db_b[4]  = {(const float*)d_in[13], (const float*)d_in[17], (const float*)d_in[21], (const float*)d_in[25]};
  const float* db_w[4]  = {(const float*)d_in[14], (const float*)d_in[18], (const float*)d_in[22], (const float*)d_in[26]};
  const float* db_cb[4] = {(const float*)d_in[15], (const float*)d_in[19], (const float*)d_in[23], (const float*)d_in[27]};
  const float* sn1_w = (const float*)d_in[28];
  const float* sn1_b = (const float*)d_in[29];
  const float* sn2_w = (const float*)d_in[30];
  const float* sn2_b = (const float*)d_in[31];
  const float* lb_w  = (const float*)d_in[32];
  const float* lb_b  = (const float*)d_in[33];
  const float* ld_w  = (const float*)d_in[34];
  const float* ld_b  = (const float*)d_in[35];
  const float* ps_w  = (const float*)d_in[36];
  const float* ps_b  = (const float*)d_in[37];
  const float* pq_w  = (const float*)d_in[38];
  const float* pq_b  = (const float*)d_in[39];
  const float* fc1_w = (const float*)d_in[40];
  const float* fc1_b = (const float*)d_in[41];
  const float* fc2_w = (const float*)d_in[42];
  const float* fc2_b = (const float*)d_in[43];

  const size_t CN  = (size_t)CC * NN;        // 294912 (per-batch C*N)
  const size_t DN  = (size_t)DC * NN;        // 589824
  const size_t SN  = (size_t)NN * NN;        // 5308416 (per-batch scores)

  float* out     = (float*)d_out;            // (B,128,48,48)
  float* new_num = out + BB * CN;
  float* new_den = new_num + BB * CN;

  float* Wsp = (float*)d_ws;
  size_t P = 0;
  auto alloc = [&](size_t n) -> float* { float* p = Wsp + P; P += (n + 63) & ~(size_t)63; return p; };

  float* xrope    = alloc(BB * CN);
  float* kbuf     = alloc(BB * CN);
  float* vbuf     = alloc(BB * CN);
  float* rbuf     = alloc(BB * CN);
  float* rec      = alloc(BB * CN);
  float* feats    = alloc(BB * DN);              // df lives here (B,256,N)
  float* bnbuf    = alloc((size_t)BB * 224 * NN);
  float* bnscale  = alloc(256);
  float* bnshift  = alloc(256);
  float* s1       = alloc((size_t)BB * 64 * NN);
  float* s2       = alloc(BB * DN);
  float* decay    = alloc(BB * DC);
  float* sfb      = alloc(BB * DN);
  float* Bf       = alloc((size_t)BB * 32 * NN);
  float* Df       = alloc((size_t)BB * 32 * NN);
  float* scores   = alloc(BB * SN);              // reused for both attentions
  float* att1     = alloc((size_t)BB * 32 * NN);
  float* combined = alloc(BB * CN);
  float* Qb       = alloc(BB * CN);
  float* K2b      = alloc(BB * CN);
  float* V2b      = alloc(BB * CN);
  float* x1       = alloc(BB * CN);
  float* h1       = alloc((size_t)BB * 512 * NN);

  auto ggrid = [](int M, int N) { return dim3((unsigned)((N + 63) / 64), (unsigned)((M + 127) / 128), BB); };

  // ---- RWKV-style branch ----
  rope_kernel<<<EWGRID(BB * 64 * NN), 0, stream>>>(x, xrope);
  gemm_wmma<0,0,EPI_NONE><<<ggrid(128, NN), 256, 0, stream>>>(
      k_w, 0, xrope, (long long)CN, kbuf, (long long)CN, k_b, nullptr, 0,
      128, NN, 128, 128, NN, NN, 1.f);
  gemm_wmma<0,0,EPI_NONE><<<ggrid(128, NN), 256, 0, stream>>>(
      v_w, 0, xrope, (long long)CN, vbuf, (long long)CN, v_b, nullptr, 0,
      128, NN, 128, 128, NN, NN, 1.f);
  gemm_wmma<0,0,EPI_SIGMOID><<<ggrid(128, NN), 256, 0, stream>>>(
      r_w, 0, x, (long long)CN, rbuf, (long long)CN, r_b, nullptr, 0,
      128, NN, 128, 128, NN, NN, 1.f);
  wkv_kernel<<<EWGRID(BB * CC * NN), 0, stream>>>(kbuf, vbuf, rbuf, u, new_num, new_den, rec);

  // ---- Dense block (4 stages of BN -> ReLU -> 3x3 conv -> concat) ----
  copy_x_to_feats<<<EWGRID(BB * CC * NN), 0, stream>>>(x, feats);
  for (int i = 0; i < 4; ++i) {
    int cin = 128 + 32 * i;
    bn_stats<<<dim3(cin), dim3(256), 0, stream>>>(feats, db_g[i], db_b[i], bnscale, bnshift);
    bn_apply_relu<<<EWGRID(BB * cin * NN), 0, stream>>>(feats, bnscale, bnshift, bnbuf, cin);
    conv3x3_dense<<<EWGRID(BB * 32 * NN), 0, stream>>>(bnbuf, db_w[i], db_cb[i], feats, cin, 128 + 32 * i);
  }

  // ---- Spiking branch: tau, sf ----
  gemm_wmma<0,0,EPI_RELU><<<ggrid(64, NN), 256, 0, stream>>>(
      sn1_w, 0, feats, (long long)DN, s1, (long long)(64 * NN), sn1_b, nullptr, 0,
      64, NN, 256, 256, NN, NN, 1.f);
  gemm_wmma<0,0,EPI_SIGMOID><<<ggrid(256, NN), 256, 0, stream>>>(
      sn2_w, 0, s1, (long long)(64 * NN), s2, (long long)DN, sn2_b, nullptr, 0,
      256, NN, 64, 64, NN, NN, 1.f);
  tau_decay<<<dim3(BB * DC), dim3(256), 0, stream>>>(s2, decay);
  spike_fuse<<<EWGRID(BB * DC * NN), 0, stream>>>(feats, decay, sfb);

  // ---- Spike attention (32-ch) ----
  gemm_wmma<0,0,EPI_NONE><<<ggrid(32, NN), 256, 0, stream>>>(
      lb_w, 0, sfb, (long long)DN, Bf, (long long)(32 * NN), lb_b, nullptr, 0,
      32, NN, 256, 256, NN, NN, 1.f);
  gemm_wmma<0,0,EPI_NONE><<<ggrid(32, NN), 256, 0, stream>>>(
      ld_w, 0, sfb, (long long)DN, Df, (long long)(32 * NN), ld_b, nullptr, 0,
      32, NN, 256, 256, NN, NN, 1.f);
  gemm_wmma<1,0,EPI_NONE><<<ggrid(NN, NN), 256, 0, stream>>>(        // Bf^T Bf
      Bf, (long long)(32 * NN), Bf, (long long)(32 * NN), scores, (long long)SN,
      nullptr, nullptr, 0, NN, NN, 32, NN, NN, NN, 1.f);
  softmax_rows<<<dim3(BB * NN), dim3(256), 0, stream>>>(scores);
  gemm_wmma<0,1,EPI_NONE><<<ggrid(32, NN), 256, 0, stream>>>(        // Df · attn^T
      Df, (long long)(32 * NN), scores, (long long)SN, att1, (long long)(32 * NN),
      nullptr, nullptr, 0, 32, NN, NN, NN, NN, NN, 1.f);
  gemm_wmma<0,0,EPI_ADD><<<ggrid(128, NN), 256, 0, stream>>>(        // combined = ps + rec
      ps_w, 0, att1, (long long)(32 * NN), combined, (long long)CN, ps_b,
      rec, (long long)CN, 128, NN, 32, 32, NN, NN, 1.f);

  // ---- Second attention (128-ch) ----
  gemm_wmma<0,0,EPI_NONE><<<ggrid(128, NN), 256, 0, stream>>>(
      pq_w, 0, combined, (long long)CN, Qb, (long long)CN, pq_b, nullptr, 0,
      128, NN, 128, 128, NN, NN, 1.f);
  gemm_wmma<0,0,EPI_NONE><<<ggrid(128, NN), 256, 0, stream>>>(
      k_w, 0, combined, (long long)CN, K2b, (long long)CN, k_b, nullptr, 0,
      128, NN, 128, 128, NN, NN, 1.f);
  gemm_wmma<0,0,EPI_NONE><<<ggrid(128, NN), 256, 0, stream>>>(
      v_w, 0, combined, (long long)CN, V2b, (long long)CN, v_b, nullptr, 0,
      128, NN, 128, 128, NN, NN, 1.f);
  gemm_wmma<1,0,EPI_NONE><<<ggrid(NN, NN), 256, 0, stream>>>(        // Q^T K / sqrt(C)
      Qb, (long long)CN, K2b, (long long)CN, scores, (long long)SN,
      nullptr, nullptr, 0, NN, NN, 128, NN, NN, NN, 1.f / sqrtf(128.f));
  softmax_rows<<<dim3(BB * NN), dim3(256), 0, stream>>>(scores);
  gemm_wmma<0,1,EPI_ADD><<<ggrid(128, NN), 256, 0, stream>>>(        // x1 = V·aw^T + x
      V2b, (long long)CN, scores, (long long)SN, x1, (long long)CN,
      nullptr, x, (long long)CN, 128, NN, NN, NN, NN, NN, 1.f);

  // ---- FFN: out = x1 + fc2(relu(fc1(x1))^2) ----
  gemm_wmma<0,0,EPI_RELUSQ><<<ggrid(512, NN), 256, 0, stream>>>(
      fc1_w, 0, x1, (long long)CN, h1, (long long)(512 * NN), fc1_b, nullptr, 0,
      512, NN, 128, 128, NN, NN, 1.f);
  gemm_wmma<0,0,EPI_ADD><<<ggrid(128, NN), 256, 0, stream>>>(
      fc2_w, 0, h1, (long long)(512 * NN), out, (long long)CN, fc2_b,
      x1, (long long)CN, 128, NN, 512, 512, NN, NN, 1.f);
}
